// MinGRUSynthetic_24653112279159
// MI455X (gfx1250) — matrix operand order repaired and checked by reading, not compiled, not run on hardware
//
#include <hip/hip_runtime.h>

typedef __bf16 bf16_t;
typedef __attribute__((ext_vector_type(16))) __bf16 v16bf;
typedef __attribute__((ext_vector_type(8)))  __bf16 v8bf;
typedef __attribute__((ext_vector_type(4)))  __bf16 v4bf;
typedef __attribute__((ext_vector_type(8)))  float  v8f;
typedef __attribute__((ext_vector_type(4)))  unsigned int u32x4;
typedef __attribute__((ext_vector_type(8)))  int i32x8;
typedef __attribute__((ext_vector_type(4)))  int i32x4;

#define BATCH   32
#define SEQ     8192
#define DIM     256
#define CLASSES 8
#define KP      264          // padded K pitch (bf16): 528B rows -> 16B aligned, bank-rotated
#define MCHUNK  64           // rows of A staged per iteration
#define NCHUNK  64           // scan chunks along S
#define CHUNK   128          // SEQ / NCHUNK

// ---------------------------------------------------------------------------
// Fused (gather +) GEMM + minGRU gate epilogue.
//   zpre = A @ Wz + bz ;  h~ = A @ Wh + bh
//   a = 1 - sigmoid(zpre) ; b = sigmoid(zpre) * h~   (both stored bf16)
// Wz/Wh live in LDS as bf16 transposed [n][k] for the block lifetime (~270KB
// of the 320KB WGP LDS). A panels (64x256) are staged through LDS: via the
// Tensor Data Mover (layer 1, bf16 source, HW row padding) or via vectorized
// gather+downconvert (layer 0, f32 embedding rows).
// ---------------------------------------------------------------------------
template<bool GATHER>
__global__ __launch_bounds__(256)
void mingru_gemm_act(const float* __restrict__ emb,
                     const int*   __restrict__ x,
                     const bf16_t* __restrict__ hIn,
                     const float* __restrict__ Wz, const float* __restrict__ bz,
                     const float* __restrict__ Wh, const float* __restrict__ bh,
                     bf16_t* __restrict__ aOut, bf16_t* __restrict__ bOut)
{
    extern __shared__ char smem[];
    bf16_t* Wzs = (bf16_t*)smem;            // [DIM][KP]  transposed: Wzs[n*KP + k]
    bf16_t* Whs = Wzs + DIM * KP;           // [DIM][KP]
    bf16_t* As  = Whs + DIM * KP;           // [MCHUNK][KP] row-major

    const int t = threadIdx.x;              // 0..255
    const int b = blockIdx.x;               // batch

    // Load + transpose + downconvert both weights: thread t owns column n = t.
    for (int k = 0; k < DIM; ++k) {
        Wzs[t * KP + k] = (bf16_t)Wz[k * DIM + t];
        Whs[t * KP + k] = (bf16_t)Wh[k * DIM + t];
    }
    __syncthreads();

    const int lane = t & 31;
    const int wid  = t >> 5;
    const int mi   = wid & 3;               // m-tile within panel (4 x 16 rows)
    const int nh   = wid >> 2;              // n half (2 x 128 cols)
    const int hi16 = (lane >> 4) & 1;       // upper half-wave?
    const int l15  = lane & 15;

    const v8f vzero = {0.f,0.f,0.f,0.f,0.f,0.f,0.f,0.f};

    for (int sbase = blockIdx.y * MCHUNK; sbase < SEQ; sbase += gridDim.y * MCHUNK) {
        __syncthreads();                    // guard A panel WAR
        if (GATHER) {
            // f32 embedding rows -> bf16 LDS: float4 loads, 4xbf16 LDS stores.
            const int cgrp  = t & 63;       // 4-column group
            const int rquad = t >> 6;       // 0..3
            #pragma unroll
            for (int i = 0; i < 16; ++i) {
                const int r   = rquad * 16 + i;
                const int row = x[b * SEQ + sbase + r];
                const float4 v = *(const float4*)(emb + (size_t)row * DIM + cgrp * 4);
                v4bf p;
                p[0] = (bf16_t)v.x; p[1] = (bf16_t)v.y;
                p[2] = (bf16_t)v.z; p[3] = (bf16_t)v.w;
                *(v4bf*)(As + r * KP + cgrp * 4) = p;
            }
        } else {
            // bf16 source: one TDM descriptor moves the whole 64x256 panel,
            // padding each 512B row by 4 DWORDs to reproduce KP=264 in LDS.
            if (t == 0) {
                const unsigned long long gaddr =
                    (unsigned long long)(const void*)(hIn + ((size_t)b * SEQ + sbase) * DIM);
                const unsigned ldsOff = (unsigned)(unsigned long long)(void*)As;
                u32x4 g0;
                g0[0] = 1u;                                          // count=1, user mode
                g0[1] = ldsOff;                                      // lds_addr
                g0[2] = (unsigned)(gaddr & 0xFFFFFFFFull);           // global_addr[31:0]
                g0[3] = (unsigned)((gaddr >> 32) & 0x01FFFFFFull)    // global_addr[56:32]
                        | (2u << 30);                                // type=2 (image)
                i32x8 g1;
                g1[0] = (1 << 16)      // data_size = 1 (2 bytes)
                      | (1 << 20)      // pad_enable
                      | (6 << 22)      // pad_interval: 128 DWORDs (one 256xbf16 row)
                      | (3 << 25);     // pad_amount: 4 DWORDs (8 bf16 -> KP=264)
                g1[1] = (int)(256u << 16);            // tensor_dim0 = 256 (lo16)
                g1[2] = (int)(64u << 16);             // tensor_dim0 hi16=0, tensor_dim1 = 64
                g1[3] = (int)(256u << 16);            // tensor_dim1 hi16=0, tile_dim0 = 256
                g1[4] = 64;                           // tile_dim1 = 64 rows, tile_dim2 = 0
                g1[5] = 256;                          // tensor_dim0_stride = 256 (lo32)
                g1[6] = 0;                            // stride hi16 / dim1_stride lo16
                g1[7] = 0;                            // dim1_stride hi32
                const i32x4 gz4 = {0, 0, 0, 0};
                const i32x8 gz8 = {0, 0, 0, 0, 0, 0, 0, 0};
                __builtin_amdgcn_tensor_load_to_lds(g0, g1, gz4, gz4, gz8, 0);
                __builtin_amdgcn_s_wait_tensorcnt(0);
            }
        }
        __syncthreads();

        v8f accZ[8], accH[8];
        #pragma unroll
        for (int i = 0; i < 8; ++i) { accZ[i] = vzero; accH[i] = vzero; }

        const int arow = mi * 16 + l15;
        #pragma unroll
        for (int k0 = 0; k0 < DIM; k0 += 32) {
            // A fragment (16x32 bf16): lanes 0-15 -> K {0..7,16..23}, lanes 16-31 -> +8
            const bf16_t* ap = As + arow * KP + k0 + hi16 * 8;
            const v8bf alo = *(const v8bf*)(ap);
            const v8bf ahi = *(const v8bf*)(ap + 16);
            v16bf afrag;
            #pragma unroll
            for (int i = 0; i < 8; ++i) { afrag[i] = alo[i]; afrag[i + 8] = ahi[i]; }

            #pragma unroll
            for (int nt = 0; nt < 8; ++nt) {
                const int n = nh * 128 + nt * 16 + l15;
                // B fragment (32x16 bf16): column n, lanes 0-15 -> K {0..7,16..23}, lanes 16-31 -> +8
                const bf16_t* bpz = Wzs + n * KP + k0 + hi16 * 8;
                const bf16_t* bph = Whs + n * KP + k0 + hi16 * 8;
                const v8bf z0 = *(const v8bf*)(bpz);
                const v8bf z1 = *(const v8bf*)(bpz + 16);
                const v8bf h0 = *(const v8bf*)(bph);
                const v8bf h1 = *(const v8bf*)(bph + 16);
                v16bf bfz, bfh;
                #pragma unroll
                for (int i = 0; i < 8; ++i) {
                    bfz[i] = z0[i]; bfz[i + 8] = z1[i];
                    bfh[i] = h0[i]; bfh[i + 8] = h1[i];
                }
                accZ[nt] = __builtin_amdgcn_wmma_f32_16x16x32_bf16(
                    false, afrag, false, bfz, (short)0, accZ[nt], false, false);
                accH[nt] = __builtin_amdgcn_wmma_f32_16x16x32_bf16(
                    false, afrag, false, bfh, (short)0, accH[nt], false, false);
            }
        }

        // Epilogue: fuse sigmoid gate, emit a = 1-z and b = z*h~ (bf16).
        // C/D layout: element j -> m = j + 8*hi16, n = l15 (within tile).
        #pragma unroll
        for (int nt = 0; nt < 8; ++nt) {
            const int n = nh * 128 + nt * 16 + l15;
            const float bzn = bz[n];
            const float bhn = bh[n];
            #pragma unroll
            for (int j = 0; j < 8; ++j) {
                const int m = mi * 16 + hi16 * 8 + j;
                const size_t off = ((size_t)b * SEQ + sbase + m) * DIM + n;
                const float zp = accZ[nt][j] + bzn;
                const float zs = 1.f / (1.f + __expf(-zp));
                const float ht = accH[nt][j] + bhn;
                aOut[off] = (bf16_t)(1.f - zs);
                bOut[off] = (bf16_t)(zs * ht);
            }
        }
    }
}

// ---------------------------------------------------------------------------
// Chunked linear-recurrence scan: h_t = a_t * h_{t-1} + b_t, h_0 = 0.
// ---------------------------------------------------------------------------
__global__ __launch_bounds__(256)
void scan_pass1(const bf16_t* __restrict__ aArr, const bf16_t* __restrict__ bArr,
                float* __restrict__ agg)   // agg[tid]=A, agg[TOT+tid]=B
{
    const int tid = blockIdx.x * 256 + threadIdx.x;   // ((b*NCHUNK + c)*DIM + d)
    const int d = tid & (DIM - 1);
    const int c = (tid >> 8) & (NCHUNK - 1);
    const int b = tid >> 14;
    const size_t base = ((size_t)b * SEQ + c * CHUNK) * DIM + d;
    const int TOT = BATCH * NCHUNK * DIM;

    float A = 1.f, Bv = 0.f;
    for (int s = 0; s < CHUNK; ++s) {
        __builtin_prefetch(aArr + base + (size_t)(s + 16) * DIM, 0, 0);
        const float av = (float)aArr[base + (size_t)s * DIM];
        const float bv = (float)bArr[base + (size_t)s * DIM];
        A *= av;
        Bv = Bv * av + bv;
    }
    agg[tid]       = A;
    agg[TOT + tid] = Bv;
}

__global__ __launch_bounds__(256)
void scan_carry(const float* __restrict__ agg, float* __restrict__ carry)
{
    const int tid = blockIdx.x * 256 + threadIdx.x;   // b*DIM + d
    const int d = tid & (DIM - 1);
    const int b = tid >> 8;
    const int TOT = BATCH * NCHUNK * DIM;

    float Bv = 0.f;                                    // h entering chunk 0 is 0
    for (int c = 0; c < NCHUNK; ++c) {
        const int idx = ((b * NCHUNK + c) << 8) | d;
        carry[idx] = Bv;
        const float Ac = agg[idx];
        const float Bc = agg[TOT + idx];
        Bv = Bv * Ac + Bc;
    }
}

__global__ __launch_bounds__(256)
void scan_pass3(const bf16_t* __restrict__ aArr, const bf16_t* __restrict__ bArr,
                const float* __restrict__ carry, bf16_t* __restrict__ hOut)
{
    const int tid = blockIdx.x * 256 + threadIdx.x;
    const int d = tid & (DIM - 1);
    const int c = (tid >> 8) & (NCHUNK - 1);
    const int b = tid >> 14;
    const size_t base = ((size_t)b * SEQ + c * CHUNK) * DIM + d;

    float h = carry[tid];
    for (int s = 0; s < CHUNK; ++s) {
        __builtin_prefetch(bArr + base + (size_t)(s + 16) * DIM, 0, 0);
        const float av = (float)aArr[base + (size_t)s * DIM];
        const float bv = (float)bArr[base + (size_t)s * DIM];
        h = av * h + bv;
        hOut[base + (size_t)s * DIM] = (bf16_t)h;
    }
}

// ---------------------------------------------------------------------------
// Classifier on last timestep: out[b,c] = h[b,S-1,:] . Wo[:,c] + bo[c]
// ---------------------------------------------------------------------------
__global__ __launch_bounds__(256)
void classifier_k(const bf16_t* __restrict__ h, const float* __restrict__ Wo,
                  const float* __restrict__ bo, float* __restrict__ out)
{
    const int t = threadIdx.x;            // 256 = 32 batches x 8 classes
    const int b = t >> 3;
    const int c = t & 7;
    float acc = bo[c];
    const bf16_t* hrow = h + ((size_t)b * SEQ + (SEQ - 1)) * DIM;
    for (int d = 0; d < DIM; ++d)
        acc += (float)hrow[d] * Wo[d * CLASSES + c];
    out[b * CLASSES + c] = acc;
}

// ---------------------------------------------------------------------------
extern "C" void kernel_launch(void* const* d_in, const int* in_sizes, int n_in,
                              void* d_out, int out_size, void* d_ws, size_t ws_size,
                              hipStream_t stream)
{
    const int*   x   = (const int*)  d_in[0];
    const float* emb = (const float*)d_in[1];
    const float* Wz  = (const float*)d_in[2];
    const float* bz  = (const float*)d_in[3];
    const float* Wh  = (const float*)d_in[4];
    const float* bh  = (const float*)d_in[5];
    const float* Wo  = (const float*)d_in[6];
    const float* bo  = (const float*)d_in[7];
    float* out = (float*)d_out;

    const size_t SZ = (size_t)BATCH * SEQ * DIM;      // 64M elems
    bf16_t* aArr = (bf16_t*)d_ws;                     // 128 MB
    bf16_t* bArr = aArr + SZ;                         // 128 MB
    bf16_t* hArr = bArr + SZ;                         // 128 MB
    float*  agg  = (float*)(hArr + SZ);               // 2 * 512K floats
    float*  carr = agg + 2 * (BATCH * NCHUNK * DIM);  // 512K floats

    const size_t shmem = (size_t)(2 * DIM * KP + MCHUNK * KP) * sizeof(bf16_t); // ~297 KB
    const dim3 gemmGrid(BATCH, 32);
    const int scanBlocks = (BATCH * NCHUNK * DIM) / 256;   // 2048

    for (int l = 0; l < 2; ++l) {
        const float* Wzl = Wz + (size_t)l * DIM * DIM;
        const float* Whl = Wh + (size_t)l * DIM * DIM;
        const float* bzl = bz + (size_t)l * DIM;
        const float* bhl = bh + (size_t)l * DIM;
        if (l == 0) {
            mingru_gemm_act<true><<<gemmGrid, 256, shmem, stream>>>(
                emb, x, hArr, Wzl, bzl, Whl, bhl, aArr, bArr);
        } else {
            mingru_gemm_act<false><<<gemmGrid, 256, shmem, stream>>>(
                emb, x, hArr, Wzl, bzl, Whl, bhl, aArr, bArr);
        }
        scan_pass1<<<scanBlocks, 256, 0, stream>>>(aArr, bArr, agg);
        scan_carry<<<(BATCH * DIM) / 256, 256, 0, stream>>>(agg, carr);
        scan_pass3<<<scanBlocks, 256, 0, stream>>>(aArr, bArr, carr, hArr);
    }
    classifier_k<<<1, 256, 0, stream>>>(hArr, Wo, bo, out);
}
